// MixtureOfExperts_1735166788700
// MI455X (gfx1250) — matrix operand order repaired and checked by reading, not compiled, not run on hardware
//
#include <hip/hip_runtime.h>
#include <hip/hip_bf16.h>

// ---------------------------------------------------------------------------
// MoE pipeline for MI455X (gfx1250, wave32, WMMA bf16 16x16x32, f32 accum)
//
//  x[8192,4096] -> 10 experts (4096->120->84->10) -> gate mix (M=10 -> N=100)
//  -> 100 task heads (10->120->84->10) -> out [100,8192,10] + logits_loss[100]
// ---------------------------------------------------------------------------

typedef __attribute__((ext_vector_type(16))) __bf16          v16bf;
typedef __attribute__((ext_vector_type(2)))  __bf16          v2bf;
typedef __attribute__((ext_vector_type(8)))  float           v8f;
typedef __attribute__((ext_vector_type(2)))  float           f2;
typedef __attribute__((ext_vector_type(4)))  float           f4;
typedef __attribute__((ext_vector_type(8)))  unsigned short  us8;
typedef __attribute__((ext_vector_type(16))) unsigned short  us16;

#define BQ   8192
#define INQ  4096

// --- f32 -> bf16: hw builtin if present, else native fptrunc (legalized) ---
__device__ __forceinline__ unsigned pkbf16(float lo, float hi) {
#if __has_builtin(__builtin_amdgcn_cvt_pk_bf16_f32)
    auto r = __builtin_amdgcn_cvt_pk_bf16_f32(lo, hi);
    return __builtin_bit_cast(unsigned, r);
#else
    v2bf t;
    t.x = (__bf16)lo;
    t.y = (__bf16)hi;
    return __builtin_bit_cast(unsigned, t);
#endif
}

__device__ __forceinline__ unsigned short f2bf(float x) {
    return (unsigned short)(pkbf16(x, x) & 0xFFFFu);
}

__device__ __forceinline__ v16bf mkfrag(us8 lo, us8 hi) {
    union { us16 v; us8 h[2]; } u;
    u.h[0] = lo; u.h[1] = hi;
    return __builtin_bit_cast(v16bf, u.v);
}

__device__ __forceinline__ v8f wmma_bf16(v16bf a, v16bf b, v8f c) {
    return __builtin_amdgcn_wmma_f32_16x16x32_bf16(
        false, a, false, b, (short)0, c, false, false);
}

// A fragment (16-bit A 16x32 ISA layout): lane holds row (lane&15),
// element e -> K = k_base + (e + (e>=8 ? 8 : 0) + 8*half)
__device__ __forceinline__ v16bf afrag_from_f32(const float* p0 /* = row + k0 + 8*half */) {
    f4 v0 = *(const f4*)(p0);
    f4 v1 = *(const f4*)(p0 + 4);
    f4 v2 = *(const f4*)(p0 + 16);
    f4 v3 = *(const f4*)(p0 + 20);
    union { us16 v; unsigned u[8]; } q;
    q.u[0] = pkbf16(v0.x, v0.y); q.u[1] = pkbf16(v0.z, v0.w);
    q.u[2] = pkbf16(v1.x, v1.y); q.u[3] = pkbf16(v1.z, v1.w);
    q.u[4] = pkbf16(v2.x, v2.y); q.u[5] = pkbf16(v2.z, v2.w);
    q.u[6] = pkbf16(v3.x, v3.y); q.u[7] = pkbf16(v3.z, v3.w);
    return __builtin_bit_cast(v16bf, q.v);
}

// A fragment from a bf16 row: p = row_base + k0 + 8*half (two b128 loads)
__device__ __forceinline__ v16bf afrag_from_bf16(const unsigned short* p) {
    return mkfrag(*(const us8*)p, *(const us8*)(p + 16));
}

// B fragment (col-major staged tile): lane holds column (lane&15),
// element e -> K = k0 + (e + (e>=8 ? 8 : 0) + 8*half) ; colbase = col*stride + k0 + 8*half
__device__ __forceinline__ v16bf bfrag_from_lds(const unsigned short* colbase) {
    return mkfrag(*(const us8*)colbase, *(const us8*)(colbase + 16));
}

// ---------------------------------------------------------------------------
// Kernel A: h1[m] = relu(x @ eW1[m] + eb1[m])   -> bf16 [10][8192][128] (120 pad)
// grid (64 row-tiles, 10 experts), 256 threads. K staged in 128-wide slabs:
// one barrier pair + one staging pass feeds 4 WMMA bursts (32 wmma/wave).
// ---------------------------------------------------------------------------
__global__ __launch_bounds__(256) void expert_fc1_kernel(
    const float* __restrict__ X, const float* __restrict__ W1,
    const float* __restrict__ b1, unsigned short* __restrict__ h1)
{
    __shared__ unsigned short Wt[128 * 128];   // [col][k] bf16, 32KB
    unsigned* Wt32 = (unsigned*)Wt;            // [col][64 k-pairs]

    const int m       = blockIdx.y;
    const int rowBase = blockIdx.x * 128;
    const int tid     = threadIdx.x;
    const int w       = tid >> 5;
    const int lane    = tid & 31;
    const int half    = lane >> 4;
    const int lr      = lane & 15;

    const float* Wm   = W1 + (size_t)m * INQ * 120;
    const int    row  = rowBase + w * 16 + lr;
    const float* xrow = X + (size_t)row * INQ;

    v8f acc[8];
#pragma unroll
    for (int i = 0; i < 8; i++) acc[i] = (v8f)0.0f;

    for (int k0 = 0; k0 < INQ; k0 += 128) {
        __syncthreads();
        // stage 128x120 slab of W1[m] transposed -> Wt[col][k]; pack k-pairs
        // into b32 LDS stores; zero-pad cols 120..127
        {
            const int c0  = (tid & 31) * 4;    // column group (float4)
            const int kp0 = tid >> 5;          // k-pair base 0..7
#pragma unroll
            for (int i = 0; i < 8; i++) {
                int kp = kp0 + i * 8;          // 0..63
                int k  = kp * 2;
                f4 va = (f4)0.0f, vb = (f4)0.0f;
                if (c0 < 120) {
                    va = *(const f4*)(Wm + (size_t)(k0 + k) * 120 + c0);
                    vb = *(const f4*)(Wm + (size_t)(k0 + k + 1) * 120 + c0);
                }
                Wt32[(c0 + 0) * 64 + kp] = pkbf16(va.x, vb.x);
                Wt32[(c0 + 1) * 64 + kp] = pkbf16(va.y, vb.y);
                Wt32[(c0 + 2) * 64 + kp] = pkbf16(va.z, vb.z);
                Wt32[(c0 + 3) * 64 + kp] = pkbf16(va.w, vb.w);
            }
        }
        __syncthreads();

#pragma unroll
        for (int ks = 0; ks < 4; ks++) {
            const int kk = ks * 32;
            v16bf bf[8];
#pragma unroll
            for (int ct = 0; ct < 8; ct++)
                bf[ct] = bfrag_from_lds(&Wt[(ct * 16 + lr) * 128 + kk + 8 * half]);
            v16bf a = afrag_from_f32(xrow + k0 + kk + 8 * half);
#pragma unroll
            for (int ct = 0; ct < 8; ct++)
                acc[ct] = wmma_bf16(a, bf[ct], acc[ct]);
        }
    }

    // bias + relu, store bf16
#pragma unroll
    for (int ct = 0; ct < 8; ct++) {
        int col = ct * 16 + lr;
        float bias = (col < 120) ? b1[m * 120 + col] : 0.0f;
#pragma unroll
        for (int i = 0; i < 8; i++) {
            int r = rowBase + w * 16 + i + 8 * half;
            float v = acc[ct][i] + bias;
            v = v > 0.0f ? v : 0.0f;
            h1[((size_t)m * BQ + r) * 128 + col] = f2bf(v);
        }
    }
}

// ---------------------------------------------------------------------------
// Kernel B (fused FC2+FC3): expert_out[m] = relu(h1 @ W2 + b2) @ W3 + b3
// h1 bf16 [10][8192][128] -> EO f32 [10][8192][16] (10 pad; pad cols exactly 0)
// ---------------------------------------------------------------------------
__global__ __launch_bounds__(256) void expert_fc23_kernel(
    const unsigned short* __restrict__ h1,
    const float* __restrict__ W2, const float* __restrict__ b2,
    const float* __restrict__ W3, const float* __restrict__ b3,
    float* __restrict__ EO)
{
    __shared__ unsigned short W2t[96 * 128];        // [col][k] 24KB
    __shared__ unsigned short W3t[16 * 96];         // [col][k] 3KB
    __shared__ unsigned short strip[8][16 * 96];    // per-wave h2 strips, 24KB

    const int m       = blockIdx.y;
    const int rowBase = blockIdx.x * 128;
    const int tid     = threadIdx.x;
    const int w       = tid >> 5;
    const int lane    = tid & 31;
    const int half    = lane >> 4;
    const int lr      = lane & 15;

    // stage W2[m] (120x84 -> 128x96, transposed, bf16) and W3[m]; b32 packed
    {
        unsigned* W2t32 = (unsigned*)W2t;           // [col][64 k-pairs]
        const float* W2m = W2 + (size_t)m * 120 * 84;
        for (int idx = tid; idx < 96 * 64; idx += 256) {
            int c = idx % 96, kp = idx / 96, k = kp * 2;
            float a = (c < 84 && k     < 120) ? W2m[(k    ) * 84 + c] : 0.0f;
            float b = (c < 84 && k + 1 < 120) ? W2m[(k + 1) * 84 + c] : 0.0f;
            W2t32[c * 64 + kp] = pkbf16(a, b);
        }
        unsigned* W3t32 = (unsigned*)W3t;           // [col][48 k-pairs]
        const float* W3m = W3 + (size_t)m * 84 * 10;
        for (int idx = tid; idx < 16 * 48; idx += 256) {
            int c = idx % 16, kp = idx / 16, k = kp * 2;
            float a = (c < 10 && k     < 84) ? W3m[(k    ) * 10 + c] : 0.0f;
            float b = (c < 10 && k + 1 < 84) ? W3m[(k + 1) * 10 + c] : 0.0f;
            W3t32[c * 48 + kp] = pkbf16(a, b);
        }
    }
    __syncthreads();

    // FC2
    v8f acc2[6];
#pragma unroll
    for (int i = 0; i < 6; i++) acc2[i] = (v8f)0.0f;

    const unsigned short* h1row =
        h1 + ((size_t)m * BQ + rowBase + w * 16 + lr) * 128;
#pragma unroll
    for (int kt = 0; kt < 4; kt++) {
        int k0 = kt * 32;
        v16bf bf[6];
#pragma unroll
        for (int ct = 0; ct < 6; ct++)
            bf[ct] = bfrag_from_lds(&W2t[(ct * 16 + lr) * 128 + k0 + 8 * half]);
        v16bf a = afrag_from_bf16(h1row + k0 + 8 * half);
#pragma unroll
        for (int ct = 0; ct < 6; ct++)
            acc2[ct] = wmma_bf16(a, bf[ct], acc2[ct]);
    }
    // bias + relu -> per-wave LDS strip [16][96]
#pragma unroll
    for (int ct = 0; ct < 6; ct++) {
        int col = ct * 16 + lr;
        float bias = (col < 84) ? b2[m * 84 + col] : 0.0f;
#pragma unroll
        for (int i = 0; i < 8; i++) {
            float v = acc2[ct][i] + bias;
            v = v > 0.0f ? v : 0.0f;
            strip[w][(i + 8 * half) * 96 + col] = f2bf(v);
        }
    }
    __syncthreads();

    // FC3
    v8f acc3 = (v8f)0.0f;
    const unsigned short* srow = &strip[w][lr * 96];
#pragma unroll
    for (int kt = 0; kt < 3; kt++) {
        int k0 = kt * 32;
        v16bf a = afrag_from_bf16(srow + k0 + 8 * half);
        v16bf b = bfrag_from_lds(&W3t[lr * 96 + k0 + 8 * half]);
        acc3 = wmma_bf16(a, b, acc3);
    }
    {
        int col = lr;
        float bias = (col < 10) ? b3[m * 10 + col] : 0.0f;
#pragma unroll
        for (int i = 0; i < 8; i++) {
            int r = rowBase + w * 16 + i + 8 * half;
            EO[((size_t)m * BQ + r) * 16 + col] = acc3[i] + bias;
        }
    }
}

// ---------------------------------------------------------------------------
// Kernel C (fused gate mix + full task head per (n, row-tile)):
//  G = sum_m wg[m,n]*EO[m]  (VALU, -> LDS bf16 [128][32], K pad 10->32)
//  t = relu(G@tW1+b) ; t = relu(t@tW2+b) ; out = t@tW3 + b  -> [n][8192][10]
// ---------------------------------------------------------------------------
__global__ __launch_bounds__(256) void task_head_kernel(
    const float* __restrict__ EO,
    const float* __restrict__ gate_w, const int* __restrict__ gate_mask,
    const float* __restrict__ tW1, const float* __restrict__ tb1,
    const float* __restrict__ tW2, const float* __restrict__ tb2,
    const float* __restrict__ tW3, const float* __restrict__ tb3,
    float* __restrict__ out)
{
    __shared__ unsigned short Gt[128 * 32];        // [row][k]   8KB
    __shared__ unsigned short W1t[128 * 32];       // [col][k]   8KB
    __shared__ unsigned short W2t[96 * 128];       // [col][k]  24KB
    __shared__ unsigned short W3t[16 * 96];        // [col][k]   3KB
    __shared__ unsigned short strip1[8][16 * 128]; // t1 strips 32KB
    __shared__ unsigned short strip2[8][16 * 96];  // t2 strips 24KB

    const int n       = blockIdx.y;
    const int rowBase = blockIdx.x * 128;
    const int tid     = threadIdx.x;
    const int w       = tid >> 5;
    const int lane    = tid & 31;
    const int half    = lane >> 4;
    const int lr      = lane & 15;

    // effective gate weights for this task head
    float wg[10];
#pragma unroll
    for (int mm = 0; mm < 10; mm++)
        wg[mm] = gate_w[mm * 100 + n] * (float)gate_mask[mm * 100 + n];

    // G tile (gate-mixed expert outputs), bf16, K padded to 32; b32 packed
    {
        unsigned* Gt32 = (unsigned*)Gt;            // [row][16 f-pairs]
        for (int idx = tid; idx < 128 * 16; idx += 256) {
            int r = idx >> 4, fp = idx & 15;
            float g0 = 0.0f, g1 = 0.0f;
            if (fp < 5) {                          // f = 2*fp, 2*fp+1 < 10
                int row = rowBase + r;
#pragma unroll
                for (int mm = 0; mm < 10; mm++) {
                    f2 e = *(const f2*)(EO + ((size_t)mm * BQ + row) * 16 + fp * 2);
                    g0 += wg[mm] * e.x;
                    g1 += wg[mm] * e.y;
                }
            }
            Gt32[r * 16 + fp] = pkbf16(g0, g1);
        }
    }
    // stage tW1[n] (10x120 -> 32x128 transposed), tW2[n], tW3[n]; b32 packed
    {
        unsigned* W1t32 = (unsigned*)W1t;          // [col][16 k-pairs]
        const float* W1n = tW1 + (size_t)n * 10 * 120;
        for (int idx = tid; idx < 128 * 16; idx += 256) {
            int kp = idx >> 7, c = idx & 127, k = kp * 2;
            float a = (c < 120 && k     < 10) ? W1n[(k    ) * 120 + c] : 0.0f;
            float b = (c < 120 && k + 1 < 10) ? W1n[(k + 1) * 120 + c] : 0.0f;
            W1t32[c * 16 + kp] = pkbf16(a, b);
        }
        unsigned* W2t32 = (unsigned*)W2t;          // [col][64 k-pairs]
        const float* W2n = tW2 + (size_t)n * 120 * 84;
        for (int idx = tid; idx < 96 * 64; idx += 256) {
            int c = idx % 96, kp = idx / 96, k = kp * 2;
            float a = (c < 84 && k     < 120) ? W2n[(k    ) * 84 + c] : 0.0f;
            float b = (c < 84 && k + 1 < 120) ? W2n[(k + 1) * 84 + c] : 0.0f;
            W2t32[c * 64 + kp] = pkbf16(a, b);
        }
        unsigned* W3t32 = (unsigned*)W3t;          // [col][48 k-pairs]
        const float* W3n = tW3 + (size_t)n * 84 * 10;
        for (int idx = tid; idx < 16 * 48; idx += 256) {
            int c = idx % 16, kp = idx / 16, k = kp * 2;
            float a = (c < 10 && k     < 84) ? W3n[(k    ) * 10 + c] : 0.0f;
            float b = (c < 10 && k + 1 < 84) ? W3n[(k + 1) * 10 + c] : 0.0f;
            W3t32[c * 48 + kp] = pkbf16(a, b);
        }
    }
    __syncthreads();

    // FC1: single K-step (K=32)
    v8f acc1[8];
#pragma unroll
    for (int i = 0; i < 8; i++) acc1[i] = (v8f)0.0f;
    {
        v16bf bf[8];
#pragma unroll
        for (int ct = 0; ct < 8; ct++)
            bf[ct] = bfrag_from_lds(&W1t[(ct * 16 + lr) * 32 + 8 * half]);
        v16bf a = afrag_from_bf16(&Gt[(w * 16 + lr) * 32 + 8 * half]);
#pragma unroll
        for (int ct = 0; ct < 8; ct++)
            acc1[ct] = wmma_bf16(a, bf[ct], acc1[ct]);
    }
#pragma unroll
    for (int ct = 0; ct < 8; ct++) {
        int col = ct * 16 + lr;
        float bias = (col < 120) ? tb1[n * 120 + col] : 0.0f;
#pragma unroll
        for (int i = 0; i < 8; i++) {
            float v = acc1[ct][i] + bias;
            v = v > 0.0f ? v : 0.0f;
            strip1[w][(i + 8 * half) * 128 + col] = f2bf(v);
        }
    }
    __syncthreads();

    // FC2: K = 128 (120 pad), N = 96 (84 pad)
    v8f acc2[6];
#pragma unroll
    for (int i = 0; i < 6; i++) acc2[i] = (v8f)0.0f;
    {
        const unsigned short* s1 = &strip1[w][lr * 128];
#pragma unroll
        for (int kt = 0; kt < 4; kt++) {
            int k0 = kt * 32;
            v16bf bf[6];
#pragma unroll
            for (int ct = 0; ct < 6; ct++)
                bf[ct] = bfrag_from_lds(&W2t[(ct * 16 + lr) * 128 + k0 + 8 * half]);
            v16bf a = afrag_from_bf16(s1 + k0 + 8 * half);
#pragma unroll
            for (int ct = 0; ct < 6; ct++)
                acc2[ct] = wmma_bf16(a, bf[ct], acc2[ct]);
        }
    }
#pragma unroll
    for (int ct = 0; ct < 6; ct++) {
        int col = ct * 16 + lr;
        float bias = (col < 84) ? tb2[n * 84 + col] : 0.0f;
#pragma unroll
        for (int i = 0; i < 8; i++) {
            float v = acc2[ct][i] + bias;
            v = v > 0.0f ? v : 0.0f;
            strip2[w][(i + 8 * half) * 96 + col] = f2bf(v);
        }
    }
    __syncthreads();

    // FC3: K = 96 (84 pad), N = 16 (10 pad); store only cols < 10
    v8f acc3 = (v8f)0.0f;
    {
        const unsigned short* s2 = &strip2[w][lr * 96];
#pragma unroll
        for (int kt = 0; kt < 3; kt++) {
            int k0 = kt * 32;
            v16bf a = afrag_from_bf16(s2 + k0 + 8 * half);
            v16bf b = bfrag_from_lds(&W3t[lr * 96 + k0 + 8 * half]);
            acc3 = wmma_bf16(a, b, acc3);
        }
    }
    if (lr < 10) {
        float bias = tb3[n * 10 + lr];
#pragma unroll
        for (int i = 0; i < 8; i++) {
            int r = rowBase + w * 16 + i + 8 * half;
            out[((size_t)n * BQ + r) * 10 + lr] = acc3[i] + bias;
        }
    }
}

// ---------------------------------------------------------------------------
// Kernel D: logits_loss[n] = sum_m log_sigmoid(gate_logits[m,n])
// ---------------------------------------------------------------------------
__global__ void logits_loss_kernel(const float* __restrict__ gl,
                                   float* __restrict__ out)
{
    int n = blockIdx.x * blockDim.x + threadIdx.x;
    if (n < 100) {
        float s = 0.0f;
#pragma unroll
        for (int m = 0; m < 10; m++) {
            float x  = gl[m * 100 + n];
            float mn = x < 0.0f ? x : 0.0f;
            s += mn - log1pf(__expf(-fabsf(x)));
        }
        out[n] = s;
    }
}

// ---------------------------------------------------------------------------
extern "C" void kernel_launch(void* const* d_in, const int* in_sizes, int n_in,
                              void* d_out, int out_size, void* d_ws, size_t ws_size,
                              hipStream_t stream)
{
    (void)in_sizes; (void)n_in; (void)out_size; (void)ws_size;

    const float* x    = (const float*)d_in[0];
    // d_in[1] = task (unused by the reference math)
    const float* eW1  = (const float*)d_in[2];
    const float* eb1  = (const float*)d_in[3];
    const float* eW2  = (const float*)d_in[4];
    const float* eb2  = (const float*)d_in[5];
    const float* eW3  = (const float*)d_in[6];
    const float* eb3  = (const float*)d_in[7];
    const float* gw   = (const float*)d_in[8];
    const float* gl   = (const float*)d_in[9];
    const int*   gm   = (const int*)d_in[10];
    const float* tW1  = (const float*)d_in[11];
    const float* tb1  = (const float*)d_in[12];
    const float* tW2  = (const float*)d_in[13];
    const float* tb2  = (const float*)d_in[14];
    const float* tW3  = (const float*)d_in[15];
    const float* tb3  = (const float*)d_in[16];

    // workspace layout: h1 bf16 [10][8192][128], then EO f32 [10][8192][16]
    unsigned short* h1 = (unsigned short*)d_ws;
    float* EO = (float*)((char*)d_ws + (size_t)10 * BQ * 128 * sizeof(unsigned short));

    float* out = (float*)d_out;
    float* out_logits = out + (size_t)100 * BQ * 10;

    expert_fc1_kernel <<<dim3(BQ / 128, 10),  256, 0, stream>>>(x, eW1, eb1, h1);
    expert_fc23_kernel<<<dim3(BQ / 128, 10),  256, 0, stream>>>(h1, eW2, eb2, eW3, eb3, EO);
    task_head_kernel  <<<dim3(BQ / 128, 100), 256, 0, stream>>>(EO, gw, gm,
                                                                tW1, tb1, tW2, tb2, tW3, tb3, out);
    logits_loss_kernel<<<1, 128, 0, stream>>>(gl, out_logits);
}
